// GATEncoder_24721831755890
// MI455X (gfx1250) — matrix operand order repaired and checked by reading, not compiled, not run on hardware
//
#include <hip/hip_runtime.h>
#include <math.h>

#define NNODES 50000
#define NPAD   50016      // multiple of 32 (50016 = 32*1563) for WMMA tiling
#define NEDGES 800000
#define ETOT   850000     // + self loops
#define FDIM   256

typedef __bf16 bf16;
typedef __attribute__((ext_vector_type(16))) __bf16 v16bf;
typedef __attribute__((ext_vector_type(8)))  __bf16 v8bf;
typedef __attribute__((ext_vector_type(8)))  float  v8f;
typedef __attribute__((ext_vector_type(4)))  unsigned v4u;
typedef __attribute__((ext_vector_type(8)))  int      v8i;
typedef __attribute__((ext_vector_type(4)))  int      v4i;

// ---------- helpers ----------
__device__ __forceinline__ bf16 f2bf(float f) {
  unsigned u = __float_as_uint(f);
  u += 0x7FFFu + ((u >> 16) & 1u);            // RNE
  unsigned short h = (unsigned short)(u >> 16);
  return __builtin_bit_cast(bf16, h);
}
// order-preserving float<->uint key (for atomicMax-based segment max)
__device__ __forceinline__ unsigned f2key(float f) {
  unsigned b = __float_as_uint(f);
  return (b & 0x80000000u) ? ~b : (b | 0x80000000u);
}
__device__ __forceinline__ float key2f(unsigned k) {
  unsigned b = (k & 0x80000000u) ? (k & 0x7FFFFFFFu) : ~k;
  return __uint_as_float(b);
}

// ---------- pack W[256,256] f32 -> bf16 in WMMA B-fragment friendly layout ----------
// Wpk[(kb*256 + n)*32 + kk] = W[(kb*32+kk)*256 + n], kb=0..7, kk=0..31
__global__ void k_pack_w(const float* __restrict__ W, bf16* __restrict__ Wpk) {
  int i  = blockIdx.x * blockDim.x + threadIdx.x;   // 0..65535
  int kk = i & 31;
  int n  = (i >> 5) & 255;
  int kb = i >> 13;
  Wpk[i] = f2bf(W[(kb * 32 + kk) * FDIM + n]);
}

// ---------- convert activations f32 -> bf16 row-major, zero pad rows ----------
__global__ void k_to_bf16(const float* __restrict__ in, bf16* __restrict__ out) {
  int i = blockIdx.x * blockDim.x + threadIdx.x;    // NPAD*256 threads
  int row = i >> 8;
  float v = (row < NNODES) ? in[i] : 0.0f;
  out[i] = f2bf(v);
}

// ---------- GEMM: XH[NPAD,256] = Abf[NPAD,256] @ W (via packed Wpk) ----------
// Block = 128 threads (4 waves), covers a 32-row strip of A staged in LDS by
// the Tensor Data Mover; each wave computes a 32x64 output tile (64 WMMAs).
// A strip lives in DYNAMIC shared memory (16 KB, group-segment offset 0) so
// the optimizer cannot prove it unwritten and elide the ds_loads (the TDM
// write is invisible to alias analysis).
__global__ void __launch_bounds__(128)
k_gemm(const bf16* __restrict__ A, const bf16* __restrict__ Bpk, float* __restrict__ C) {
  extern __shared__ bf16 smA[];                     // 32*FDIM bf16 = 16 KB at LDS offset 0

  const int lane = threadIdx.x & 31;
  const int wv   = threadIdx.x >> 5;
  const int hf   = lane >> 4;        // lane half (0/1)
  const int r    = lane & 15;
  const int m0   = blockIdx.x * 32;
  const int n0   = wv * 64;

  // ---- TDM: DMA the 32x256 bf16 A strip into LDS (wave 0 only) ----
  if (threadIdx.x < 32) {
    unsigned long long ga = (unsigned long long)A + (unsigned long long)m0 * (FDIM * 2);
    // D# group0: count=1 | lds_addr=0 | global_addr | type=2
    v4u g0 = { 1u,
               0u,
               (unsigned)(ga & 0xFFFFFFFFu),
               (unsigned)((ga >> 32) & 0x01FFFFFFu) | 0x80000000u };
    // D# group1 (bit layout per ISA 8.4):
    //  d0: wg_mask=0, data_size=1 (2 bytes)
    //  d1[31:16] = tensor_dim0[15:0] = 256
    //  d2[31:16] = tensor_dim1[15:0] = NPAD
    //  d3[31:16] = tile_dim0 = 256
    //  d4[15:0]  = tile_dim1 = 32 ; tile_dim2 = 0
    //  d5        = tensor_dim0_stride = 256
    v8i g1 = { (int)(1u << 16),
               (int)((unsigned)FDIM << 16),
               (int)(((unsigned)NPAD & 0xFFFFu) << 16),
               (int)((((unsigned)NPAD >> 16) & 0xFFFFu) | ((unsigned)FDIM << 16)),
               32,
               FDIM,
               0, 0 };
    v4i gz4 = { 0, 0, 0, 0 };
    v8i gz8 = { 0, 0, 0, 0, 0, 0, 0, 0 };
    __builtin_amdgcn_tensor_load_to_lds(g0, g1, gz4, gz4, gz8, 0);
    __builtin_amdgcn_s_wait_tensorcnt(0);
  }
  __syncthreads();

  v8f acc00 = {}, acc01 = {}, acc02 = {}, acc03 = {};
  v8f acc10 = {}, acc11 = {}, acc12 = {}, acc13 = {};

  for (int kb = 0; kb < 8; ++kb) {
    // A fragments from LDS (16 contiguous K per lane half, matching B), rows r and 16+r
    union { v16bf v; v8bf h[2]; } a0, a1;
    const bf16* ar0 = smA + (size_t)r        * FDIM + kb * 32;
    const bf16* ar1 = smA + (size_t)(16 + r) * FDIM + kb * 32;
    a0.h[0] = *(const v8bf*)(ar0 + hf * 16);
    a0.h[1] = *(const v8bf*)(ar0 + hf * 16 + 8);
    a1.h[0] = *(const v8bf*)(ar1 + hf * 16);
    a1.h[1] = *(const v8bf*)(ar1 + hf * 16 + 8);

    // B fragments: 16 contiguous K per lane from pre-packed W (L2 resident)
    const bf16* bb = Bpk + ((size_t)kb * 256) * 32 + (size_t)hf * 16;
    v16bf b0 = *(const v16bf*)(bb + (size_t)(n0 +  0 + r) * 32);
    v16bf b1 = *(const v16bf*)(bb + (size_t)(n0 + 16 + r) * 32);
    v16bf b2 = *(const v16bf*)(bb + (size_t)(n0 + 32 + r) * 32);
    v16bf b3 = *(const v16bf*)(bb + (size_t)(n0 + 48 + r) * 32);

    acc00 = __builtin_amdgcn_wmma_f32_16x16x32_bf16(false, a0.v, false, b0, (short)0, acc00, false, false);
    acc10 = __builtin_amdgcn_wmma_f32_16x16x32_bf16(false, a1.v, false, b0, (short)0, acc10, false, false);
    acc01 = __builtin_amdgcn_wmma_f32_16x16x32_bf16(false, a0.v, false, b1, (short)0, acc01, false, false);
    acc11 = __builtin_amdgcn_wmma_f32_16x16x32_bf16(false, a1.v, false, b1, (short)0, acc11, false, false);
    acc02 = __builtin_amdgcn_wmma_f32_16x16x32_bf16(false, a0.v, false, b2, (short)0, acc02, false, false);
    acc12 = __builtin_amdgcn_wmma_f32_16x16x32_bf16(false, a1.v, false, b2, (short)0, acc12, false, false);
    acc03 = __builtin_amdgcn_wmma_f32_16x16x32_bf16(false, a0.v, false, b3, (short)0, acc03, false, false);
    acc13 = __builtin_amdgcn_wmma_f32_16x16x32_bf16(false, a1.v, false, b3, (short)0, acc13, false, false);
  }

  // C layout: VGPR i -> row base + i + hf*8, col = n-base + r
  for (int i = 0; i < 8; ++i) {
    size_t row0 = (size_t)(m0      + i + hf * 8) * FDIM;
    size_t row1 = (size_t)(m0 + 16 + i + hf * 8) * FDIM;
    C[row0 + n0 +  0 + r] = acc00[i];
    C[row0 + n0 + 16 + r] = acc01[i];
    C[row0 + n0 + 32 + r] = acc02[i];
    C[row0 + n0 + 48 + r] = acc03[i];
    C[row1 + n0 +  0 + r] = acc10[i];
    C[row1 + n0 + 16 + r] = acc11[i];
    C[row1 + n0 + 32 + r] = acc12[i];
    C[row1 + n0 + 48 + r] = acc13[i];
  }
}

// ---------- per-node attention dots: a_s[n,h], a_d[n,h] ----------
__global__ void k_attn(const float* __restrict__ XH, const float* __restrict__ att_s,
                       const float* __restrict__ att_d, float* __restrict__ as_o,
                       float* __restrict__ ad_o, int H, int Cc) {
  int i = blockIdx.x * blockDim.x + threadIdx.x;
  if (i >= NNODES * H) return;
  int n = i / H, h = i - n * H;
  const float* xr = XH + (size_t)n * FDIM + h * Cc;
  const float* sv = att_s + h * Cc;
  const float* dv = att_d + h * Cc;
  float s = 0.f, d = 0.f;
  for (int c = 0; c < Cc; ++c) { float xv = xr[c]; s += xv * sv[c]; d += xv * dv[c]; }
  as_o[i] = s; ad_o[i] = d;
}

// ---------- init accumulators ----------
__global__ void k_init(unsigned* __restrict__ maxk, float* __restrict__ denom,
                       float* __restrict__ acc, int H) {
  int i = blockIdx.x * blockDim.x + threadIdx.x;     // NNODES*FDIM threads
  if (i < NNODES * H) { maxk[i] = 0u; denom[i] = 0.f; }
  if (i < NNODES * FDIM) acc[i] = 0.f;
}

// ---------- edge pass 1: logits + segment max ----------
__global__ void k_edge_logits(const int* __restrict__ ei, const float* __restrict__ as_,
                              const float* __restrict__ ad_, float* __restrict__ ebuf,
                              unsigned* __restrict__ maxk, int H) {
  int e = blockIdx.x * blockDim.x + threadIdx.x;
  if (e >= ETOT) return;
  int s, d;
  if (e < NEDGES) { s = ei[e]; d = ei[NEDGES + e]; } else { s = d = e - NEDGES; }
  for (int h = 0; h < H; ++h) {
    float v = as_[s * H + h] + ad_[d * H + h];
    v = v > 0.f ? v : 0.2f * v;                       // LeakyReLU(0.2)
    ebuf[(size_t)e * H + h] = v;
    atomicMax(&maxk[d * H + h], f2key(v));
  }
}

// ---------- edge pass 2: exp(e - max) + segment sum ----------
__global__ void k_edge_exp(const int* __restrict__ ei, float* __restrict__ ebuf,
                           const unsigned* __restrict__ maxk, float* __restrict__ denom, int H) {
  int e = blockIdx.x * blockDim.x + threadIdx.x;
  if (e >= ETOT) return;
  int d;
  if (e < NEDGES) { d = ei[NEDGES + e]; } else { d = e - NEDGES; }
  for (int h = 0; h < H; ++h) {
    float m = key2f(maxk[d * H + h]);
    float p = __expf(ebuf[(size_t)e * H + h] - m);
    ebuf[(size_t)e * H + h] = p;
    atomicAdd(&denom[d * H + h], p);
  }
}

// ---------- edge pass 3: weighted message scatter (one wave per edge) ----------
__global__ void __launch_bounds__(256)
k_msg(const int* __restrict__ ei, const float* __restrict__ ebuf,
      const float* __restrict__ denom, const float* __restrict__ XH,
      float* __restrict__ acc, int H) {
  int lane = threadIdx.x & 31;
  int e = blockIdx.x * 8 + (threadIdx.x >> 5);
  if (e >= ETOT) return;
  int s, d;
  if (e < NEDGES) { s = ei[e]; d = ei[NEDGES + e]; } else { s = d = e - NEDGES; }
  float alpha[4];
  for (int h = 0; h < H; ++h)
    alpha[h] = ebuf[(size_t)e * H + h] / (denom[d * H + h] + 1e-16f);
  const float* xs = XH + (size_t)s * FDIM;
  float* od = acc + (size_t)d * FDIM;
  int cshift = (H == 4) ? 6 : 8;                      // channel -> head
  for (int j = 0; j < 8; ++j) {
    int c = j * 32 + lane;
    atomicAdd(&od[c], alpha[c >> cshift] * xs[c]);
  }
}

// ---------- finalize: + bias, optional ELU (in place) ----------
__global__ void k_final(float* __restrict__ acc, const float* __restrict__ bias, int do_elu) {
  int i = blockIdx.x * blockDim.x + threadIdx.x;
  if (i >= NNODES * FDIM) return;
  float v = acc[i] + bias[i & 255];
  if (do_elu) v = v > 0.f ? v : expm1f(v);
  acc[i] = v;
}

// ---------- launch ----------
extern "C" void kernel_launch(void* const* d_in, const int* in_sizes, int n_in,
                              void* d_out, int out_size, void* d_ws, size_t ws_size,
                              hipStream_t stream) {
  (void)in_sizes; (void)n_in; (void)out_size; (void)ws_size;
  const float* x  = (const float*)d_in[0];
  const int*   ei = (const int*)d_in[1];
  const float* Wt[3] = {(const float*)d_in[2], (const float*)d_in[6],  (const float*)d_in[10]};
  const float* As[3] = {(const float*)d_in[3], (const float*)d_in[7],  (const float*)d_in[11]};
  const float* Ad[3] = {(const float*)d_in[4], (const float*)d_in[8],  (const float*)d_in[12]};
  const float* Bi[3] = {(const float*)d_in[5], (const float*)d_in[9],  (const float*)d_in[13]};
  float* out = (float*)d_out;

  char* w = (char*)d_ws;
  auto carve = [&](size_t bytes) -> void* {
    void* p = (void*)w; w += (bytes + 255) & ~(size_t)255; return p;
  };
  float*    XH    = (float*)   carve((size_t)NPAD * FDIM * 4);
  float*    P     = (float*)   carve((size_t)NPAD * FDIM * 4);
  bf16*     xbf   = (bf16*)    carve((size_t)NPAD * FDIM * 2);
  bf16*     Wpk   = (bf16*)    carve((size_t)FDIM * FDIM * 2);
  float*    asb   = (float*)   carve((size_t)NNODES * 4 * 4);
  float*    adb   = (float*)   carve((size_t)NNODES * 4 * 4);
  unsigned* maxk  = (unsigned*)carve((size_t)NNODES * 4 * 4);
  float*    denom = (float*)   carve((size_t)NNODES * 4 * 4);
  float*    ebuf  = (float*)   carve((size_t)ETOT * 4 * 4);

  const int H[3]   = {4, 4, 1};
  const int Cc[3]  = {64, 64, 256};
  const int ELU[3] = {1, 1, 0};
  const float* IN[3] = {x, P, P};
  float* ACC[3]      = {P, P, out};

  const int TB = 256;
  const size_t gemm_lds = (size_t)32 * FDIM * sizeof(bf16);   // 16 KB dynamic LDS
  for (int l = 0; l < 3; ++l) {
    // pack weights, convert activations to bf16 (padded)
    k_pack_w <<<(FDIM * FDIM) / TB, TB, 0, stream>>>(Wt[l], Wpk);
    k_to_bf16<<<((size_t)NPAD * FDIM) / TB, TB, 0, stream>>>(IN[l], xbf);
    // dense projection via bf16 WMMA (A strip staged in LDS by TDM)
    k_gemm   <<<NPAD / 32, 128, gemm_lds, stream>>>(xbf, Wpk, XH);
    // attention scalars
    int nh = NNODES * H[l];
    k_attn   <<<(nh + TB - 1) / TB, TB, 0, stream>>>(XH, As[l], Ad[l], asb, adb, H[l], Cc[l]);
    // accumulators (safe: GEMM has already consumed IN[l] == ACC[l] for l==1)
    k_init   <<<(NNODES * FDIM) / TB, TB, 0, stream>>>(maxk, denom, ACC[l], H[l]);
    // segment softmax + aggregation
    int eb = (ETOT + TB - 1) / TB;
    k_edge_logits<<<eb, TB, 0, stream>>>(ei, asb, adb, ebuf, maxk, H[l]);
    k_edge_exp   <<<eb, TB, 0, stream>>>(ei, ebuf, maxk, denom, H[l]);
    k_msg        <<<(ETOT + 7) / 8, TB, 0, stream>>>(ei, ebuf, denom, XH, ACC[l], H[l]);
    // bias + activation
    k_final      <<<(NNODES * FDIM) / TB, TB, 0, stream>>>(ACC[l], Bi[l], ELU[l]);
  }
}
